// MiniQwen3NextSparseMoeBlock_74517682586453
// MI455X (gfx1250) — matrix operand order repaired and checked by reading, not compiled, not run on hardware
//
#include <hip/hip_runtime.h>

// MiniQwen3Next Sparse MoE block for gfx1250 (MI455X), bf16 WMMA path, v3b.
// v3b: tensor_load_to_lds uses the 6-arg (clang-23 / amdgpu-toolchain) form.

#define DEVINL __device__ __forceinline__

constexpr int Hdim = 2048;
constexpr int Tn   = 2048;    // B*S tokens
constexpr int En   = 16;
constexpr int TOPK = 4;
constexpr int IM   = 512;     // I_MOE
constexpr int ISH  = 4096;    // I_SH
constexpr int KMOE = En * IM; // 8192

constexpr int TILE_M = 32;    // tokens per block (2 m-tiles of 16)
constexpr int TILE_K = 64;    // K elements staged per chunk

typedef __attribute__((ext_vector_type(16))) __bf16         v16bf;
typedef __attribute__((ext_vector_type(8)))  float          v8f;
typedef __attribute__((ext_vector_type(4)))  unsigned short v4u;
typedef __attribute__((ext_vector_type(4)))  unsigned int   u32x4;
typedef __attribute__((ext_vector_type(8)))  int            i32x8;
typedef __attribute__((ext_vector_type(4)))  int            i32x4;

union Frag { v16bf v; unsigned short u[16]; };

DEVINL unsigned short f2bf(float f) {
  union { __bf16 b; unsigned short u; } c;
  c.b = (__bf16)f;
  return c.u;
}

DEVINL float fast_rcp(float x)  { return __builtin_amdgcn_rcpf(x); }
DEVINL float sigmoidf_(float x) { return fast_rcp(1.f + __expf(-x)); }
DEVINL float siluf_(float x)    { return x * sigmoidf_(x); }

// A fragment (16x32 bf16) from an LDS tile [TILE_M][TILE_K], row r.
// lane&15 = M row, kh = lane>>4; elems 0..7 -> K=kk+8*kh+0..7, 8..15 -> +16.
DEVINL v16bf load_a_lds(const unsigned short* __restrict__ tile, int r, int kk,
                        int kh) {
  const unsigned short* p0 = tile + r * TILE_K + kk + kh * 8;
  const unsigned short* p1 = p0 + 16;
  Frag f;
#pragma unroll
  for (int j = 0; j < 8; ++j) { f.u[j] = p0[j]; f.u[8 + j] = p1[j]; }
  return f.v;
}

// B fragment (32x16 bf16): lane&15 = N col, kh = lane>>4;
// elems 0..7 -> K = kb + 8*kh + 0..7, 8..15 -> +16 (W[out][in]).
DEVINL v16bf load_b_bf16(const unsigned short* __restrict__ row, int kb, int kh) {
  const unsigned short* p0 = row + kb + kh * 8;
  const unsigned short* p1 = p0 + 16;
  Frag f;
#pragma unroll
  for (int j = 0; j < 8; ++j) { f.u[j] = p0[j]; f.u[8 + j] = p1[j]; }
  return f.v;
}

DEVINL v8f wmma_bf16(v16bf a, v16bf b, v8f c) {
  return __builtin_amdgcn_wmma_f32_16x16x32_bf16(false, a, false, b, (short)0, c,
                                                 false, false);
}

// ---------------------------------------------------------------------------
// TDM: issue an async load of a [TILE_M x TILE_K] bf16 tile from a row-major
// tensor (row length = rowlen elements) into LDS at ldsoff. 2D D# per ISA 8.3/8.4.
// ---------------------------------------------------------------------------
DEVINL void tdm_load_a(const unsigned short* gsrc, unsigned ldsoff,
                       unsigned rowlen) {
  unsigned long long ga = (unsigned long long)(size_t)gsrc;
  u32x4 g0;
  g0[0] = 1u;                                           // count=1 (valid user D#)
  g0[1] = ldsoff;                                       // lds_addr [63:32]
  g0[2] = (unsigned)(ga & 0xffffffffu);                 // global_addr[31:0]
  g0[3] = (unsigned)((ga >> 32) & 0x01ffffffu) | 0x80000000u; // addr[56:32]|type=2
  i32x8 g1;
  g1[0] = 0x00010000;                       // wg_mask=0, data_size=1 (2 bytes)
  g1[1] = (int)((rowlen & 0xffffu) << 16);  // tensor_dim0[15:0]  (bits 63:48)
  g1[2] = (int)((rowlen >> 16) | ((unsigned)(Tn & 0xffff) << 16)); // dim0 hi|dim1 lo
  g1[3] = (int)(TILE_K << 16);              // dim1 hi | tile_dim0 = 64
  g1[4] = TILE_M;                           // tile_dim1 = 32 (tile_dim2 = 0)
  g1[5] = (int)rowlen;                      // tensor_dim0_stride[31:0]
  g1[6] = 0;                                // stride hi | dim1_stride lo
  g1[7] = 0;
  i32x4 z4 = {0, 0, 0, 0};                  // groups 2/3 unused for 2D tile
  i32x8 z8 = {0, 0, 0, 0, 0, 0, 0, 0};
  __builtin_amdgcn_tensor_load_to_lds(g0, g1, z4, z4, z8, 0);
}

// ---------------------------------------------------------------------------
// fp32 -> bf16 streaming convert: 4 elements/thread.
// ---------------------------------------------------------------------------
__global__ void cvt_bf16_kernel(const float* __restrict__ src,
                                unsigned short* __restrict__ dst, int n4) {
  int i = blockIdx.x * blockDim.x + threadIdx.x;
  if (i >= n4) return;
  float4 f = ((const float4*)src)[i];
  v4u o;
  o.x = f2bf(f.x); o.y = f2bf(f.y); o.z = f2bf(f.z); o.w = f2bf(f.w);
  ((v4u*)dst)[i] = o;
}

// ---------------------------------------------------------------------------
// Router: one block (8 waves) per token; thread 0 softmax/top-4/renorm.
// ---------------------------------------------------------------------------
__global__ void router_kernel(const float* __restrict__ x,
                              const float* __restrict__ rw,
                              const float* __restrict__ sgw,
                              float* __restrict__ combine,
                              float* __restrict__ sgate,
                              float* __restrict__ logits_out) {
  const int t    = blockIdx.x;
  const int lane = threadIdx.x & 31;   // wave32
  const int wave = threadIdx.x >> 5;
  const float* xt = x + (size_t)t * Hdim;
  const float* r0 = rw + (size_t)(2 * wave) * Hdim;
  const float* r1 = rw + (size_t)(2 * wave + 1) * Hdim;

  __shared__ float s_log[En];
  __shared__ float s_g;

  float a0 = 0.f, a1 = 0.f, ag = 0.f;
  for (int h = lane; h < Hdim; h += 32) {
    float xv = xt[h];
    a0 += xv * r0[h];
    a1 += xv * r1[h];
    if (wave == 0) ag += xv * sgw[h];
  }
#pragma unroll
  for (int off = 16; off > 0; off >>= 1) {
    a0 += __shfl_xor(a0, off, 32);
    a1 += __shfl_xor(a1, off, 32);
    ag += __shfl_xor(ag, off, 32);
  }
  if (lane == 0) {
    s_log[2 * wave]     = a0;
    s_log[2 * wave + 1] = a1;
    if (wave == 0) s_g = ag;
  }
  __syncthreads();

  if (threadIdx.x == 0) {
    float lg[En], pv[En];
    float m = s_log[0];
#pragma unroll
    for (int e = 0; e < En; ++e) { lg[e] = s_log[e]; m = fmaxf(m, lg[e]); }
    float sum = 0.f;
#pragma unroll
    for (int e = 0; e < En; ++e) { pv[e] = __expf(lg[e] - m); sum += pv[e]; }
    float inv = fast_rcp(sum);
#pragma unroll
    for (int e = 0; e < En; ++e) pv[e] *= inv;

    int   sel[TOPK];
    float sw[TOPK];
    float csum = 0.f;
#pragma unroll
    for (int k = 0; k < TOPK; ++k) {
      int am = 0; float mv = -1.f;
#pragma unroll
      for (int e = 0; e < En; ++e)
        if (pv[e] > mv) { mv = pv[e]; am = e; }
      sel[k] = am; sw[k] = mv; csum += mv; pv[am] = -1.f;
    }
    float rinv = fast_rcp(csum);
    float* cb = combine + (size_t)t * En;
#pragma unroll
    for (int e = 0; e < En; ++e) cb[e] = 0.f;
#pragma unroll
    for (int k = 0; k < TOPK; ++k) cb[sel[k]] = sw[k] * rinv;

    sgate[t] = sigmoidf_(s_g);
    float* lo = logits_out + (size_t)t * En;
#pragma unroll
    for (int e = 0; e < En; ++e) lo[e] = lg[e];
  }
}

// Double-buffer pattern in all GEMM kernels: wave 0 prefetches chunk c+1 via
// TDM, waits TENSORcnt (1 mid-stream, 0 on last), barrier, compute, barrier.

// ---------------------------------------------------------------------------
// Shared expert gate+up: grid(T/32, ISH/128). Epilogue silu(g)*u -> act_sh.
// ---------------------------------------------------------------------------
__global__ void sh_gateup_kernel(const unsigned short* __restrict__ bx,
                                 const unsigned short* __restrict__ wg,
                                 const unsigned short* __restrict__ wu,
                                 unsigned short* __restrict__ act_sh) {
  __shared__ __align__(16) unsigned short sA[2][TILE_M * TILE_K];
  const int lane = threadIdx.x & 31;
  const int wave = threadIdx.x >> 5;
  const int lm   = lane & 15;
  const int kh   = lane >> 4;
  const int t0   = blockIdx.x * TILE_M;
  const int n0   = (blockIdx.y * 8 + wave) * 16;

  const unsigned short* atile = bx + (size_t)t0 * Hdim;
  const unsigned short* bgrow = wg + (size_t)(n0 + lm) * Hdim;
  const unsigned short* burow = wu + (size_t)(n0 + lm) * Hdim;
  const unsigned lds0 = (unsigned)(size_t)&sA[0][0];
  const unsigned lds1 = (unsigned)(size_t)&sA[1][0];

  v8f g0 = {0,0,0,0,0,0,0,0}, g1 = {0,0,0,0,0,0,0,0};
  v8f u0 = {0,0,0,0,0,0,0,0}, u1 = {0,0,0,0,0,0,0,0};

  constexpr int NC = Hdim / TILE_K;
  if (wave == 0) tdm_load_a(atile, lds0, Hdim);
  for (int c = 0; c < NC; ++c) {
    const int kb = c * TILE_K;
    if (wave == 0) {
      if (c + 1 < NC) {
        tdm_load_a(atile + (kb + TILE_K), (c & 1) ? lds0 : lds1, Hdim);
        __builtin_amdgcn_s_wait_tensorcnt(1);
      } else {
        __builtin_amdgcn_s_wait_tensorcnt(0);
      }
    }
    __syncthreads();
    const unsigned short* tile = &sA[c & 1][0];
#pragma unroll
    for (int s = 0; s < 2; ++s) {
      const int kk = s * 32;
      v16bf a0 = load_a_lds(tile, lm, kk, kh);
      v16bf a1 = load_a_lds(tile, 16 + lm, kk, kh);
      v16bf fg = load_b_bf16(bgrow, kb + kk, kh);
      v16bf fu = load_b_bf16(burow, kb + kk, kh);
      g0 = wmma_bf16(a0, fg, g0);
      g1 = wmma_bf16(a1, fg, g1);
      u0 = wmma_bf16(a0, fu, u0);
      u1 = wmma_bf16(a1, fu, u1);
    }
    __syncthreads();
  }

#pragma unroll
  for (int v = 0; v < 8; ++v) {
    int tt = t0 + v + 8 * kh;                 // M = v + 8*kh, N = lane&15
    act_sh[(size_t)tt * ISH + (n0 + lm)]        = f2bf(siluf_(g0[v]) * u0[v]);
    act_sh[(size_t)(tt + 16) * ISH + (n0 + lm)] = f2bf(siluf_(g1[v]) * u1[v]);
  }
}

// ---------------------------------------------------------------------------
// Shared expert down + sigmoid-gate: grid(T/32, H/128).
// ---------------------------------------------------------------------------
__global__ void sh_down_kernel(const unsigned short* __restrict__ act_sh,
                               const unsigned short* __restrict__ wdown, // [H,ISH]
                               const float* __restrict__ sgate,
                               float* __restrict__ out) {
  __shared__ __align__(16) unsigned short sA[2][TILE_M * TILE_K];
  const int lane = threadIdx.x & 31;
  const int wave = threadIdx.x >> 5;
  const int lm   = lane & 15;
  const int kh   = lane >> 4;
  const int t0   = blockIdx.x * TILE_M;
  const int n0   = (blockIdx.y * 8 + wave) * 16;

  const unsigned short* atile = act_sh + (size_t)t0 * ISH;
  const unsigned short* brow  = wdown + (size_t)(n0 + lm) * ISH;
  const unsigned lds0 = (unsigned)(size_t)&sA[0][0];
  const unsigned lds1 = (unsigned)(size_t)&sA[1][0];

  v8f c0 = {0,0,0,0,0,0,0,0}, c1 = {0,0,0,0,0,0,0,0};

  constexpr int NC = ISH / TILE_K;
  if (wave == 0) tdm_load_a(atile, lds0, ISH);
  for (int c = 0; c < NC; ++c) {
    const int kb = c * TILE_K;
    if (wave == 0) {
      if (c + 1 < NC) {
        tdm_load_a(atile + (kb + TILE_K), (c & 1) ? lds0 : lds1, ISH);
        __builtin_amdgcn_s_wait_tensorcnt(1);
      } else {
        __builtin_amdgcn_s_wait_tensorcnt(0);
      }
    }
    __syncthreads();
    const unsigned short* tile = &sA[c & 1][0];
#pragma unroll
    for (int s = 0; s < 2; ++s) {
      const int kk = s * 32;
      v16bf a0 = load_a_lds(tile, lm, kk, kh);
      v16bf a1 = load_a_lds(tile, 16 + lm, kk, kh);
      v16bf b  = load_b_bf16(brow, kb + kk, kh);
      c0 = wmma_bf16(a0, b, c0);
      c1 = wmma_bf16(a1, b, c1);
    }
    __syncthreads();
  }

#pragma unroll
  for (int v = 0; v < 8; ++v) {
    int tt = t0 + v + 8 * kh;
    out[(size_t)tt * Hdim + (n0 + lm)]        = sgate[tt] * c0[v];
    out[(size_t)(tt + 16) * Hdim + (n0 + lm)] = sgate[tt + 16] * c1[v];
  }
}

// ---------------------------------------------------------------------------
// MoE gate+up per expert: grid(T/32, E * IM/128). Folds combine[t,e].
// ---------------------------------------------------------------------------
__global__ void moe_gateup_kernel(const unsigned short* __restrict__ bx,
                                  const unsigned short* __restrict__ gup, // [E,2IM,H]
                                  const float* __restrict__ combine,
                                  unsigned short* __restrict__ act_moe) {
  __shared__ __align__(16) unsigned short sA[2][TILE_M * TILE_K];
  const int lane = threadIdx.x & 31;
  const int wave = threadIdx.x >> 5;
  const int lm   = lane & 15;
  const int kh   = lane >> 4;
  const int t0   = blockIdx.x * TILE_M;
  const int e    = blockIdx.y >> 2;     // IM/128 = 4 y-blocks per expert
  const int ib   = blockIdx.y & 3;
  const int i0   = (ib * 8 + wave) * 16;

  const unsigned short* atile = bx + (size_t)t0 * Hdim;
  const unsigned short* we    = gup + (size_t)e * (2 * IM) * Hdim;
  const unsigned short* bgrow = we + (size_t)(i0 + lm) * Hdim;
  const unsigned short* burow = we + (size_t)(IM + i0 + lm) * Hdim;
  const unsigned lds0 = (unsigned)(size_t)&sA[0][0];
  const unsigned lds1 = (unsigned)(size_t)&sA[1][0];

  v8f g0 = {0,0,0,0,0,0,0,0}, g1 = {0,0,0,0,0,0,0,0};
  v8f u0 = {0,0,0,0,0,0,0,0}, u1 = {0,0,0,0,0,0,0,0};

  constexpr int NC = Hdim / TILE_K;
  if (wave == 0) tdm_load_a(atile, lds0, Hdim);
  for (int c = 0; c < NC; ++c) {
    const int kb = c * TILE_K;
    if (wave == 0) {
      if (c + 1 < NC) {
        tdm_load_a(atile + (kb + TILE_K), (c & 1) ? lds0 : lds1, Hdim);
        __builtin_amdgcn_s_wait_tensorcnt(1);
      } else {
        __builtin_amdgcn_s_wait_tensorcnt(0);
      }
    }
    __syncthreads();
    const unsigned short* tile = &sA[c & 1][0];
#pragma unroll
    for (int s = 0; s < 2; ++s) {
      const int kk = s * 32;
      v16bf a0 = load_a_lds(tile, lm, kk, kh);
      v16bf a1 = load_a_lds(tile, 16 + lm, kk, kh);
      v16bf fg = load_b_bf16(bgrow, kb + kk, kh);
      v16bf fu = load_b_bf16(burow, kb + kk, kh);
      g0 = wmma_bf16(a0, fg, g0);
      g1 = wmma_bf16(a1, fg, g1);
      u0 = wmma_bf16(a0, fu, u0);
      u1 = wmma_bf16(a1, fu, u1);
    }
    __syncthreads();
  }

#pragma unroll
  for (int v = 0; v < 8; ++v) {
    int tt0 = t0 + v + 8 * kh;
    int tt1 = tt0 + 16;
    float cw0 = combine[(size_t)tt0 * En + e];  // zero unless top-4
    float cw1 = combine[(size_t)tt1 * En + e];
    act_moe[(size_t)tt0 * KMOE + (size_t)e * IM + (i0 + lm)] =
        f2bf(cw0 * siluf_(g0[v]) * u0[v]);
    act_moe[(size_t)tt1 * KMOE + (size_t)e * IM + (i0 + lm)] =
        f2bf(cw1 * siluf_(g1[v]) * u1[v]);
  }
}

// ---------------------------------------------------------------------------
// MoE down fused over all experts (K = E*IM): out[t,h] += sum act*down.
// grid(T/32, H/128).
// ---------------------------------------------------------------------------
__global__ void moe_down_kernel(const unsigned short* __restrict__ act_moe,
                                const unsigned short* __restrict__ down, // [E,H,IM]
                                float* __restrict__ out) {
  __shared__ __align__(16) unsigned short sA[2][TILE_M * TILE_K];
  const int lane = threadIdx.x & 31;
  const int wave = threadIdx.x >> 5;
  const int lm   = lane & 15;
  const int kh   = lane >> 4;
  const int t0   = blockIdx.x * TILE_M;
  const int n0   = (blockIdx.y * 8 + wave) * 16;

  const unsigned short* atile = act_moe + (size_t)t0 * KMOE;
  const unsigned short* bbase = down + (size_t)(n0 + lm) * IM;
  const unsigned lds0 = (unsigned)(size_t)&sA[0][0];
  const unsigned lds1 = (unsigned)(size_t)&sA[1][0];

  v8f c0 = {0,0,0,0,0,0,0,0}, c1 = {0,0,0,0,0,0,0,0};

  constexpr int NC = KMOE / TILE_K;   // 128 chunks; 64 | 512 so no straddle
  if (wave == 0) tdm_load_a(atile, lds0, KMOE);
  for (int c = 0; c < NC; ++c) {
    const int kb = c * TILE_K;
    if (wave == 0) {
      if (c + 1 < NC) {
        tdm_load_a(atile + (kb + TILE_K), (c & 1) ? lds0 : lds1, KMOE);
        __builtin_amdgcn_s_wait_tensorcnt(1);
      } else {
        __builtin_amdgcn_s_wait_tensorcnt(0);
      }
    }
    __syncthreads();
    const unsigned short* tile = &sA[c & 1][0];
#pragma unroll
    for (int s = 0; s < 2; ++s) {
      const int kbs = kb + s * 32;
      const int e   = kbs >> 9;        // IM = 512
      const int ik  = kbs & (IM - 1);
      const unsigned short* brow = bbase + (size_t)e * Hdim * IM + ik;
      v16bf a0 = load_a_lds(tile, lm, s * 32, kh);
      v16bf a1 = load_a_lds(tile, 16 + lm, s * 32, kh);
      v16bf b  = load_b_bf16(brow, 0, kh);
      c0 = wmma_bf16(a0, b, c0);
      c1 = wmma_bf16(a1, b, c1);
    }
    __syncthreads();
  }

#pragma unroll
  for (int v = 0; v < 8; ++v) {
    int tt = t0 + v + 8 * kh;
    size_t i0 = (size_t)tt * Hdim + (n0 + lm);
    size_t i1 = (size_t)(tt + 16) * Hdim + (n0 + lm);
    out[i0] += c0[v];   // shared_out already there
    out[i1] += c1[v];
  }
}

// ---------------------------------------------------------------------------
// Launch
// ---------------------------------------------------------------------------
extern "C" void kernel_launch(void* const* d_in, const int* in_sizes, int n_in,
                              void* d_out, int out_size, void* d_ws, size_t ws_size,
                              hipStream_t stream) {
  const float* x   = (const float*)d_in[0];
  const float* rw  = (const float*)d_in[1];
  const float* gup = (const float*)d_in[2];
  const float* dwn = (const float*)d_in[3];
  const float* shg = (const float*)d_in[4];
  const float* shu = (const float*)d_in[5];
  const float* shd = (const float*)d_in[6];
  const float* sgw = (const float*)d_in[7];

  float* out    = (float*)d_out;
  float* logits = out + (size_t)Tn * Hdim;

  char* ws = (char*)d_ws;
  size_t off = 0;
  auto take = [&](size_t bytes) { char* p = ws + off; off += (bytes + 255) & ~(size_t)255; return p; };
  float*          combine = (float*)take((size_t)Tn * En * 4);
  float*          sgate   = (float*)take((size_t)Tn * 4);
  unsigned short* bx      = (unsigned short*)take((size_t)Tn * Hdim * 2);
  unsigned short* bshg    = (unsigned short*)take((size_t)ISH * Hdim * 2);
  unsigned short* bshu    = (unsigned short*)take((size_t)ISH * Hdim * 2);
  unsigned short* bshd    = (unsigned short*)take((size_t)Hdim * ISH * 2);
  unsigned short* bgup    = (unsigned short*)take((size_t)En * 2 * IM * Hdim * 2);
  unsigned short* bdwn    = (unsigned short*)take((size_t)En * Hdim * IM * 2);
  unsigned short* act_sh  = (unsigned short*)take((size_t)Tn * ISH * 2);
  unsigned short* act_moe = (unsigned short*)take((size_t)Tn * KMOE * 2);

  auto cvt = [&](const float* s, unsigned short* d, size_t n) {
    int n4 = (int)(n / 4);
    cvt_bf16_kernel<<<(n4 + 255) / 256, 256, 0, stream>>>(s, d, n4);
  };
  cvt(x,   bx,   (size_t)Tn * Hdim);
  cvt(shg, bshg, (size_t)ISH * Hdim);
  cvt(shu, bshu, (size_t)ISH * Hdim);
  cvt(shd, bshd, (size_t)Hdim * ISH);
  cvt(gup, bgup, (size_t)En * 2 * IM * Hdim);
  cvt(dwn, bdwn, (size_t)En * Hdim * IM);

  router_kernel<<<Tn, 256, 0, stream>>>(x, rw, sgw, combine, sgate, logits);
  sh_gateup_kernel<<<dim3(Tn / TILE_M, ISH / 128), 256, 0, stream>>>(bx, bshg, bshu, act_sh);
  sh_down_kernel<<<dim3(Tn / TILE_M, Hdim / 128), 256, 0, stream>>>(act_sh, bshd, sgate, out);
  moe_gateup_kernel<<<dim3(Tn / TILE_M, En * (IM / 128)), 256, 0, stream>>>(bx, bgup, combine, act_moe);
  moe_down_kernel<<<dim3(Tn / TILE_M, Hdim / 128), 256, 0, stream>>>(act_moe, bdwn, out);
}
